// ConditionalFeedForward_28286654611535
// MI455X (gfx1250) — matrix operand order repaired and checked by reading, not compiled, not run on hardware
//
#include <hip/hip_runtime.h>

#define T_TOK 4096
#define NEXP 8
#define DDIM 1024
#define HDIM 4096
#define TOPK 2
#define NPAIR (T_TOK * TOPK)       // 8192 routed (token, expert) pairs
#define MTILE 32
#define NTILES_MAX (NPAIR / MTILE) // 256 tiles worst case per expert

typedef __attribute__((ext_vector_type(16))) __bf16 v16bf;
typedef __attribute__((ext_vector_type(8)))  float  v8f;

union FragAB {
    unsigned int u[8];
    uint4        q[2];
    v16bf        v;
};
union Acc {
    v8f   v;
    float f[8];
};

__device__ __forceinline__ unsigned short f32_to_bf16(float f) {
    return __builtin_bit_cast(unsigned short, (__bf16)f);   // native cvt, RNE
}
__device__ __forceinline__ unsigned int pack_bf16x2(float lo, float hi) {
    unsigned short l = f32_to_bf16(lo);
    unsigned short h = f32_to_bf16(hi);
    return (unsigned int)l | ((unsigned int)h << 16);
}
__device__ __forceinline__ float fast_silu(float v) {
    // v * sigmoid(v); v_exp_f32 + v_rcp_f32, no IEEE-divide expansion.
    return v * __builtin_amdgcn_rcpf(1.0f + __expf(-v));
}

#define WMMA_BF16(A, B, C) __builtin_amdgcn_wmma_f32_16x16x32_bf16( \
    false, (A), false, (B), (short)0, (C), false, false)

// ---------------- routing ----------------
__global__ void init_counts_kernel(int* counts) {
    if (threadIdx.x < NEXP) counts[threadIdx.x] = 0;
}

__global__ void route_kernel(const long long* __restrict__ idx,
                             int* __restrict__ counts,
                             int* __restrict__ lists) {
    int p = blockIdx.x * blockDim.x + threadIdx.x;
    if (p >= NPAIR) return;
    int e = (int)idx[p];                       // 0..7
    int slot = atomicAdd(&counts[e], 1);
    lists[e * NPAIR + slot] = p;
}

// ---------------- weight f32 -> bf16 pre-pass (streaming) ----------------
__global__ __launch_bounds__(256)
void convert_bf16_kernel(const float* __restrict__ src,
                         unsigned short* __restrict__ dst,
                         long n)                      // n multiple of 8
{
    long i = ((long)blockIdx.x * 256 + (long)threadIdx.x) * 8;
    if (i >= n) return;
    float4 a = *(const float4*)(src + i);
    float4 b = *(const float4*)(src + i + 4);
    uint4 o;
    o.x = pack_bf16x2(a.x, a.y);
    o.y = pack_bf16x2(a.z, a.w);
    o.z = pack_bf16x2(b.x, b.y);
    o.w = pack_bf16x2(b.z, b.w);
    *(uint4*)(dst + i) = o;
}

// ---------------- fused SwiGLU expert GEMM, 32-row M tiles ----------------
// PRECONV=true : weights already bf16 in wb pointers (raw b128 fragment loads)
// PRECONV=false: weights f32 in wf pointers (load float4 + native cvt)
template <bool PRECONV>
__global__ __launch_bounds__(256)
void moe_swiglu_kernel(const float* __restrict__ x,
                       const float* __restrict__ w1f,
                       const float* __restrict__ w2f,
                       const float* __restrict__ w3f,
                       const unsigned short* __restrict__ w1b,
                       const unsigned short* __restrict__ w2b,
                       const unsigned short* __restrict__ w3b,
                       const int*   __restrict__ counts,
                       const int*   __restrict__ lists,
                       float* __restrict__ out)        // (NPAIR, D)
{
    __shared__ unsigned short Xs[MTILE * DDIM];   // 64 KB  bf16 X tile
    __shared__ unsigned short Act[MTILE * HDIM];  // 256 KB bf16 activation tile

    const int e    = blockIdx.x / NTILES_MAX;
    const int tile = blockIdx.x % NTILES_MAX;
    const int cnt  = counts[e];
    const int base = tile * MTILE;
    if (base >= cnt) return;

    const int tid  = threadIdx.x;
    const int lane = tid & 31;
    const int wave = tid >> 5;                 // 0..7
    const int* listE = lists + e * NPAIR;

    // ---- stage 1: X rows -> LDS (bf16); 8 threads stage one row ----
    {
        int m     = tid >> 3;                  // 0..31
        int cbase = (tid & 7) * 128;           // 0..896
        int rr    = base + m;
        int tok   = (rr < cnt) ? (listE[rr] >> 1) : 0;   // p = t*TOPK + k
        const float* xr = x + (long)tok * DDIM + cbase;
        unsigned int* dst = (unsigned int*)&Xs[m * DDIM + cbase];
#pragma unroll 8
        for (int j = 0; j < 128; j += 4) {
            float4 f = *(const float4*)(xr + j);
            dst[(j >> 1)]     = pack_bf16x2(f.x, f.y);
            dst[(j >> 1) + 1] = pack_bf16x2(f.z, f.w);
        }
    }
    __syncthreads();

    // fragment lane geometry (wave32, 16x16x32 bf16)
    const int mrow  = lane & 15;               // A/C row within half
    const int ncol  = lane & 15;               // B/C column
    const int aoff  = (lane >= 16) ? 16 : 0;   // A: K 0-15 vs 16-31
    const int koffB = (lane >= 16) ? 16 : 0;   // B: K 0-15 vs 16-31
    const int mhi   = (lane >= 16) ? 8  : 0;   // C: M = r or r+8

    // ---- stage 2: act = silu(X w1^T) * (X w3^T) ; waves split H ----
#pragma unroll 1
    for (int nt = 0; nt < 32; ++nt) {
        const int hbase = wave * 512 + nt * 16;
        const long rowoff = (long)e * HDIM * DDIM + (long)(hbase + ncol) * DDIM;
        Acc a1lo, a1hi, a3lo, a3hi;
        a1lo.v = (v8f){0.f,0.f,0.f,0.f,0.f,0.f,0.f,0.f};
        a1hi.v = a1lo.v; a3lo.v = a1lo.v; a3hi.v = a1lo.v;

#pragma unroll 2
        for (int kk = 0; kk < DDIM; kk += 32) {
            FragAB alo, ahi;
            const unsigned short* xr = &Xs[mrow * DDIM + kk + aoff];
            alo.q[0] = *(const uint4*)(xr);
            alo.q[1] = *(const uint4*)(xr + 8);
            const unsigned short* xh = xr + 16 * DDIM;
            ahi.q[0] = *(const uint4*)(xh);
            ahi.q[1] = *(const uint4*)(xh + 8);

            FragAB b1, b3;
            if (PRECONV) {
                const unsigned short* p1 = w1b + rowoff + kk + koffB;
                const unsigned short* p3 = w3b + rowoff + kk + koffB;
                b1.q[0] = *(const uint4*)(p1);
                b1.q[1] = *(const uint4*)(p1 + 8);
                b3.q[0] = *(const uint4*)(p3);
                b3.q[1] = *(const uint4*)(p3 + 8);
            } else {
                const float* p1 = w1f + rowoff + kk + koffB;
                const float* p3 = w3f + rowoff + kk + koffB;
#pragma unroll
                for (int q = 0; q < 4; ++q) {
                    float4 f1 = ((const float4*)p1)[q];
                    float4 f3 = ((const float4*)p3)[q];
                    b1.u[2*q]   = pack_bf16x2(f1.x, f1.y);
                    b1.u[2*q+1] = pack_bf16x2(f1.z, f1.w);
                    b3.u[2*q]   = pack_bf16x2(f3.x, f3.y);
                    b3.u[2*q+1] = pack_bf16x2(f3.z, f3.w);
                }
            }
            // each B fragment feeds two M-halves (register-level weight reuse)
            a1lo.v = WMMA_BF16(alo.v, b1.v, a1lo.v);
            a1hi.v = WMMA_BF16(ahi.v, b1.v, a1hi.v);
            a3lo.v = WMMA_BF16(alo.v, b3.v, a3lo.v);
            a3hi.v = WMMA_BF16(ahi.v, b3.v, a3hi.v);
        }

#pragma unroll
        for (int r = 0; r < 8; ++r) {
            float glo = fast_silu(a1lo.f[r]) * a3lo.f[r];
            float ghi = fast_silu(a1hi.f[r]) * a3hi.f[r];
            Act[(r + mhi)      * HDIM + hbase + ncol] = f32_to_bf16(glo);
            Act[(16 + r + mhi) * HDIM + hbase + ncol] = f32_to_bf16(ghi);
        }
    }
    __syncthreads();

    // per-lane output row ids for the C-tile rows this lane stores (-1 = pad)
    int prow[16];
#pragma unroll
    for (int mt = 0; mt < 2; ++mt)
#pragma unroll
        for (int r = 0; r < 8; ++r) {
            int M  = mt * 16 + r + mhi;
            int rr = base + M;
            prow[mt * 8 + r] = (rr < cnt) ? listE[rr] : -1;
        }

    // ---- stage 3: out = act w2^T ; waves split D ----
#pragma unroll 1
    for (int nt = 0; nt < 8; ++nt) {
        const int dbase = wave * 128 + nt * 16;
        const long rowoff = (long)e * DDIM * HDIM + (long)(dbase + ncol) * HDIM;
        Acc aclo, achi;
        aclo.v = (v8f){0.f,0.f,0.f,0.f,0.f,0.f,0.f,0.f};
        achi.v = aclo.v;

#pragma unroll 2
        for (int kk = 0; kk < HDIM; kk += 32) {
            FragAB alo, ahi;
            const unsigned short* ar = &Act[mrow * HDIM + kk + aoff];
            alo.q[0] = *(const uint4*)(ar);
            alo.q[1] = *(const uint4*)(ar + 8);
            const unsigned short* ah = ar + 16 * HDIM;
            ahi.q[0] = *(const uint4*)(ah);
            ahi.q[1] = *(const uint4*)(ah + 8);

            FragAB b;
            if (PRECONV) {
                const unsigned short* p2 = w2b + rowoff + kk + koffB;
                b.q[0] = *(const uint4*)(p2);
                b.q[1] = *(const uint4*)(p2 + 8);
            } else {
                const float* p2 = w2f + rowoff + kk + koffB;
#pragma unroll
                for (int q = 0; q < 4; ++q) {
                    float4 f = ((const float4*)p2)[q];
                    b.u[2*q]   = pack_bf16x2(f.x, f.y);
                    b.u[2*q+1] = pack_bf16x2(f.z, f.w);
                }
            }
            aclo.v = WMMA_BF16(alo.v, b.v, aclo.v);
            achi.v = WMMA_BF16(ahi.v, b.v, achi.v);
        }

#pragma unroll
        for (int r = 0; r < 8; ++r) {
            if (prow[r] >= 0)
                out[(long)prow[r] * DDIM + dbase + ncol] = aclo.f[r];
            if (prow[8 + r] >= 0)
                out[(long)prow[8 + r] * DDIM + dbase + ncol] = achi.f[r];
        }
    }
}

extern "C" void kernel_launch(void* const* d_in, const int* in_sizes, int n_in,
                              void* d_out, int out_size, void* d_ws, size_t ws_size,
                              hipStream_t stream) {
    const float*     x   = (const float*)d_in[0];
    const long long* idx = (const long long*)d_in[1];  // int64 expert indices
    const float*     w1  = (const float*)d_in[2];
    const float*     w2  = (const float*)d_in[3];
    const float*     w3  = (const float*)d_in[4];
    float*           out = (float*)d_out;

    const long WELEMS = (long)NEXP * HDIM * DDIM;      // 33,554,432 per matrix

    // workspace layout:
    //   [0, 256)             counts (padded)
    //   [256, 256 + 256 KB)  per-expert pair lists (E * NPAIR ints)
    //   [1 MB, 1 MB + 192MB) bf16 weights (w1b, w3b, w2b; 64 MB each) if it fits
    char* wsb = (char*)d_ws;
    int* counts = (int*)wsb;
    int* lists  = (int*)(wsb + 256);
    const size_t conv_off  = (size_t)1 << 20;
    const size_t wbytes    = (size_t)WELEMS * 2;       // 64 MB
    const size_t need      = conv_off + 3 * wbytes;

    init_counts_kernel<<<1, 32, 0, stream>>>(counts);
    route_kernel<<<NPAIR / 256, 256, 0, stream>>>(idx, counts, lists);

    if (ws_size >= need) {
        unsigned short* w1b = (unsigned short*)(wsb + conv_off);
        unsigned short* w3b = w1b + WELEMS;
        unsigned short* w2b = w3b + WELEMS;
        const int cgrid = (int)(WELEMS / (8 * 256));   // 16384 blocks
        convert_bf16_kernel<<<cgrid, 256, 0, stream>>>(w1, w1b, WELEMS);
        convert_bf16_kernel<<<cgrid, 256, 0, stream>>>(w3, w3b, WELEMS);
        convert_bf16_kernel<<<cgrid, 256, 0, stream>>>(w2, w2b, WELEMS);
        moe_swiglu_kernel<true><<<NEXP * NTILES_MAX, 256, 0, stream>>>(
            x, nullptr, nullptr, nullptr, w1b, w2b, w3b, counts, lists, out);
    } else {
        moe_swiglu_kernel<false><<<NEXP * NTILES_MAX, 256, 0, stream>>>(
            x, w1, w2, w3, nullptr, nullptr, nullptr, counts, lists, out);
    }
}
